// Forest_16398185136810
// MI455X (gfx1250) — matrix operand order repaired and checked by reading, not compiled, not run on hardware
//
#include <hip/hip_runtime.h>

typedef __attribute__((ext_vector_type(16))) _Float16 v16h;
typedef __attribute__((ext_vector_type(8)))  _Float16 v8h;
typedef __attribute__((ext_vector_type(8)))  float    v8f;

#define B_TOT   16384
#define F_DIM   128
#define T_TREES 50
#define LEAVES  64
#define NSPLIT  63
#define BTILE   128
#define WAVES   8
// padded strides (break 256B-stride LDS bank conflicts; keep 16B alignment: 136 halfs = 272B)
#define XH_STR  136   // halfs per x row
#define BT_STR  136   // halfs per mask column
#define FT_STR  68    // floats per feat row (272B)

__device__ __forceinline__ float sigmoidf(float v) {
  return 1.0f / (1.0f + __expf(-v));
}

__global__ __launch_bounds__(256, 1) void forest_kernel(
    const float* __restrict__ x, const float* __restrict__ masks,
    const float* __restrict__ pi, float* __restrict__ prob_out,
    float* __restrict__ probs_out) {
  // x-tile (f16, staged once) unioned with per-wave feat tiles (used inside t loop)
  __shared__ __align__(16) union {
    _Float16 xh[BTILE * XH_STR];          // 34816 B
    float    feat[WAVES * 16 * FT_STR];   // 34816 B
  } s_u;
  __shared__ __align__(16) _Float16 s_bt[LEAVES * BT_STR];  // mask^T tile [n][k], 17408 B

  const int tid  = threadIdx.x;
  const int wave = tid >> 5;
  const int lane = tid & 31;
  const int row0 = blockIdx.x * BTILE;

  // ---- stage x tile as f16 (row-major, padded stride) ----
  for (int i = tid; i < BTILE * F_DIM; i += 256) {
    const int r = i >> 7, c = i & 127;
    s_u.xh[r * XH_STR + c] = (_Float16)x[(size_t)(row0 + r) * F_DIM + c];
  }
  __syncthreads();

  // ---- persistent A fragments: wave owns rows wave*16..+15, full K=128 ----
  // 16-bit A 16x32 layout: lane = M%16; lanes<16: elems0-7=K0-7, elems8-15=K16-23;
  //                        lanes>=16: elems0-7=K8-15, elems8-15=K24-31.
  const int mloc = (wave << 4) + (lane & 15);
  const int kg   = (lane & 16) ? 8 : 0;
  v16h afrag[4];
#pragma unroll
  for (int kt = 0; kt < 4; ++kt) {
    const v8h* p = (const v8h*)&s_u.xh[mloc * XH_STR + kt * 32 + kg];
    const v8h lo = p[0];
    const v8h hi = p[2];  // +16 halfs
    v16h a;
#pragma unroll
    for (int e = 0; e < 8; ++e) { a[e] = lo[e]; a[e + 8] = hi[e]; }
    afrag[kt] = a;
  }
  __syncthreads();  // x-tile reads done -> union becomes feat buffer

  float* const sf  = &s_u.feat[wave * 16 * FT_STR];
  const int lr     = lane >> 1;  // local row (0..15) for tree stage
  const int half   = lane & 1;   // which half of the tree (leaf bit 5)
  const int ncol   = lane & 15;
  const int kb     = (lane & 16) ? 8 : 0;    // B frag: lanes<16 K0-7,K16-23, lanes>=16 K8-15,K24-31
  const int mbase  = (lane & 16) ? 8 : 0;    // C frag rows

  for (int t = 0; t < T_TREES; ++t) {
    // ---- stage mask[t] transposed as f16: s_bt[n][k], pad n==63 with zeros ----
    const float* mt = masks + (size_t)t * F_DIM * NSPLIT;
    for (int i = tid; i < LEAVES * F_DIM; i += 256) {
      const int n = i >> 7, f = i & 127;
      const float v = (n < NSPLIT) ? mt[f * NSPLIT + n] : 0.0f;
      s_bt[n * BT_STR + f] = (_Float16)v;
    }
    if (t + 1 < T_TREES)
      __builtin_prefetch(masks + (size_t)(t + 1) * F_DIM * NSPLIT + tid * 32, 0, 0);
    __syncthreads();

    // ---- feats tile 16(b) x 64(n) via WMMA: 4 n-tiles x 4 k-steps ----
#pragma unroll
    for (int nt = 0; nt < 4; ++nt) {
      v8f acc = {};
#pragma unroll
      for (int kt = 0; kt < 4; ++kt) {
        const v8h* q = (const v8h*)&s_bt[(nt * 16 + ncol) * BT_STR + kt * 32 + kb];
        const v8h b0 = q[0];
        const v8h b1 = q[2];
        v16h bf;
#pragma unroll
        for (int e = 0; e < 8; ++e) { bf[e] = b0[e]; bf[e + 8] = b1[e]; }
        acc = __builtin_amdgcn_wmma_f32_16x16x32_f16(
            false, afrag[kt], false, bf, (short)0, acc, false, false);
      }
#pragma unroll
      for (int r = 0; r < 8; ++r)
        sf[(mbase + r) * FT_STR + nt * 16 + ncol] = acc[r];
    }
    __syncthreads();

    // ---- soft decision tree: thread = (row lr of own wave, tree half) ----
    const float* frow = sf + lr * FT_STR;
    float mu[32];
    {
      const float d0 = sigmoidf(frow[0]);
      mu[0] = half ? (1.0f - d0) : d0;
    }
    // level k nodes (our half): (2^k - 1) + half*2^(k-1) + i ; in-place expand, i descending
#define TREE_LEVEL(K, HW)                                   \
    {                                                       \
      const int beg = (1 << (K)) - 1 + half * (HW);         \
      _Pragma("unroll")                                     \
      for (int i = (HW) - 1; i >= 0; --i) {                 \
        const float d  = sigmoidf(frow[beg + i]);           \
        const float m0 = mu[i];                             \
        mu[2 * i + 1] = m0 * (1.0f - d);                    \
        mu[2 * i]     = m0 * d;                             \
      }                                                     \
    }
    TREE_LEVEL(1, 1)
    TREE_LEVEL(2, 2)
    TREE_LEVEL(3, 4)
    TREE_LEVEL(4, 8)
    TREE_LEVEL(5, 16)
#undef TREE_LEVEL

    // ---- prob[b,t,:] = mu . pi[t]  (lane-pair reduce over the two halves) ----
    const float* pit = pi + ((size_t)t * LEAVES + half * 32) * 2;
    float p0 = 0.0f, p1 = 0.0f;
#pragma unroll
    for (int j = 0; j < 32; ++j) {
      p0 = fmaf(mu[j], pit[2 * j + 0], p0);
      p1 = fmaf(mu[j], pit[2 * j + 1], p1);
    }
    p0 += __shfl_xor(p0, 1);
    p1 += __shfl_xor(p1, 1);

    const int grow = row0 + (wave << 4) + lr;
    if (half == 0) {
      prob_out[(size_t)grow * T_TREES * 2 + t * 2 + 0] = p0;
      prob_out[(size_t)grow * T_TREES * 2 + t * 2 + 1] = p1;
    }
    // probs layout (B, L, T)
    float* po = probs_out + ((size_t)grow * LEAVES + half * 32) * T_TREES + t;
#pragma unroll
    for (int j = 0; j < 32; ++j) po[(size_t)j * T_TREES] = mu[j];

    __syncthreads();  // feat reads done before next tree overwrites feat/s_bt
  }
}

extern "C" void kernel_launch(void* const* d_in, const int* in_sizes, int n_in,
                              void* d_out, int out_size, void* d_ws, size_t ws_size,
                              hipStream_t stream) {
  (void)in_sizes; (void)n_in; (void)out_size; (void)d_ws; (void)ws_size;
  const float* x     = (const float*)d_in[0];
  const float* masks = (const float*)d_in[1];
  const float* pi    = (const float*)d_in[2];
  // d_in[3] = depth scalar (fixed at 7, hardcoded)
  float* prob  = (float*)d_out;                               // (B, T, 2)
  float* probs = prob + (size_t)B_TOT * T_TREES * 2;          // (B, L, T)
  forest_kernel<<<dim3(B_TOT / BTILE), dim3(256), 0, stream>>>(
      x, masks, pi, prob, probs);
}
